// MLALayer_79688823210387
// MI455X (gfx1250) — compile-verified, matrix operand order checked by reading, not run on hardware
//
#include <hip/hip_runtime.h>
#include <hip/hip_bf16.h>

// ---------------------------------------------------------------------------
// MLA layer for MI455X (gfx1250, wave32, WMMA).
// ~235 GFLOP vs ~150MB of HBM traffic (-> ~6us at 23.3TB/s): compute-bound,
// so all matmuls route through V_WMMA_F32_16X16X32_BF16 (f32 accumulate).
// GEMMs are LDS-staged via the gfx1250 async global->LDS path (ASYNCcnt)
// to cut L2 fragment traffic ~2.5x vs direct per-wave fragment loads.
// ---------------------------------------------------------------------------

typedef __bf16 bf16;
typedef __attribute__((ext_vector_type(8)))  __bf16 v8bf;
typedef __attribute__((ext_vector_type(16))) __bf16 v16bf;
typedef __attribute__((ext_vector_type(8)))  float  v8f;
typedef __attribute__((ext_vector_type(4)))  int    v4i;

#define WMMA_BF16(a, b, c) \
  __builtin_amdgcn_wmma_f32_16x16x32_bf16(false, (a), false, (b), (short)0, (c), false, false)

#if __has_builtin(__builtin_amdgcn_global_load_async_to_lds_b128) && \
    __has_builtin(__builtin_amdgcn_s_wait_asynccnt)
#define HAVE_ASYNC_LDS 1
#else
#define HAVE_ASYNC_LDS 0
#endif

// 16-byte global->LDS stage: async DMA on gfx1250 (tracked by ASYNCcnt),
// falls back to load+ds_store if the builtin is unavailable.
// Builtin signature (from toolchain diagnostic): (int4 AS1*, int4 AS3*, Imm, Imm).
__device__ __forceinline__ void stage_copy16(const bf16* g, bf16* l) {
#if HAVE_ASYNC_LDS
  __builtin_amdgcn_global_load_async_to_lds_b128(
      (__attribute__((address_space(1))) v4i*)g,
      (__attribute__((address_space(3))) v4i*)l, 0, 0);
#else
  *(v8bf*)l = *(const v8bf*)g;
#endif
}

// Wait for the previous stage's async deposits, then make them block-visible.
__device__ __forceinline__ void wait_stage(bool hasNext) {
#if HAVE_ASYNC_LDS
  if (hasNext) __builtin_amdgcn_s_wait_asynccnt(5);  // 5 next-stage ops in flight
  else         __builtin_amdgcn_s_wait_asynccnt(0);
#endif
  __syncthreads();
}

// Per the CDNA5 ISA 16-bit 16x32 A/B fragment layout:
// lane = {half = lane>>4, m = lane&15}; elements 0..7 hold K = half*8 + e,
// elements 8..15 hold K = 16 + half*8 + e, all for row m.  Two b128 loads.
__device__ __forceinline__ v16bf load_frag(const bf16* __restrict__ rowbase, int kk, int half) {
  v8bf lo = *(const v8bf*)(rowbase + kk + half * 8);
  v8bf hi = *(const v8bf*)(rowbase + kk + 16 + half * 8);
  v16bf r;
#pragma unroll
  for (int i = 0; i < 8; ++i) { r[i] = lo[i]; r[i + 8] = hi[i]; }
  return r;
}

// Same pattern reading a (padded) LDS tile row.
__device__ __forceinline__ v16bf load_frag_lds(const bf16* rowbase, int half) {
  v8bf lo = *(const v8bf*)(rowbase + half * 8);
  v8bf hi = *(const v8bf*)(rowbase + 16 + half * 8);
  v16bf r;
#pragma unroll
  for (int i = 0; i < 8; ++i) { r[i] = lo[i]; r[i + 8] = hi[i]; }
  return r;
}

__device__ __forceinline__ int imin(int a, int b) { return a < b ? a : b; }

// ---------------------------------------------------------------------------
// Elementwise cast f32 -> bf16
// ---------------------------------------------------------------------------
__global__ void cast_f32_bf16(const float* __restrict__ in, bf16* __restrict__ out, long n) {
  long i = (long)blockIdx.x * blockDim.x + threadIdx.x;
  if (i < n) out[i] = (bf16)in[i];
}

// ---------------------------------------------------------------------------
// Cast + transpose: W[K][N] f32 -> Wt[N][K] bf16 (all dims multiples of 32)
// ---------------------------------------------------------------------------
__global__ void cast_transpose(const float* __restrict__ W, bf16* __restrict__ Wt, int K, int N) {
  __shared__ float tile[32][33];
  int n0 = blockIdx.x * 32, k0 = blockIdx.y * 32;
  int tx = threadIdx.x, ty = threadIdx.y;  // block (32,8)
#pragma unroll
  for (int i = 0; i < 4; ++i)
    tile[ty + i * 8][tx] = W[(size_t)(k0 + ty + i * 8) * N + n0 + tx];
  __syncthreads();
#pragma unroll
  for (int i = 0; i < 4; ++i)
    Wt[(size_t)(n0 + ty + i * 8) * K + k0 + tx] = (bf16)tile[tx][ty + i * 8];
}

// ---------------------------------------------------------------------------
// LDS-staged bf16 GEMM: C[M][N] = A[M][K] * Bt[N][K]^T, f32 accumulate.
// Block = 256 threads = 8 waves (2 M x 4 N), block tile 64x256, K-step 32.
// Tiles staged global->LDS via async b128 copies, double buffered.
// LDS rows padded to 40 elements (80B) so 16-lane b128 fragment reads spread
// across bank groups.
// ---------------------------------------------------------------------------
#define LDSP 40
__global__ __launch_bounds__(256) void gemm_bf16(
    const bf16* __restrict__ A, int lda, const bf16* __restrict__ Bt, int ldb,
    float* __restrict__ Cf, bf16* __restrict__ Cb, int ldc, int M, int N, int K) {
  __shared__ __align__(16) bf16 Abuf[2][64 * LDSP];
  __shared__ __align__(16) bf16 Bbuf[2][256 * LDSP];

  int tid = threadIdx.x;
  int wave = tid >> 5, lane = tid & 31;
  int half = lane >> 4, n16 = lane & 15;
  int wm = wave & 1, wn = wave >> 1;
  int rowblk = blockIdx.x * 64;
  int colblk = blockIdx.y * 256;

  // Staging assignment: 1 A chunk + 4 B chunks of 16B per thread per K-step.
  int arow = tid >> 2;
  int aseg = (tid & 3) * 8;
  const bf16* agsrc = A + (size_t)imin(rowblk + arow, M - 1) * lda + aseg;
  int aoff = arow * LDSP + aseg;
  const bf16* bgsrc[4];
  int boff[4];
#pragma unroll
  for (int j = 0; j < 4; ++j) {
    int c = tid + 256 * j;
    int brow = c >> 2;
    int bseg = (c & 3) * 8;
    bgsrc[j] = Bt + (size_t)imin(colblk + brow, N - 1) * ldb + bseg;
    boff[j] = brow * LDSP + bseg;
  }

  auto issue = [&](int kk, int st) {
    stage_copy16(agsrc + kk, &Abuf[st][aoff]);
#pragma unroll
    for (int j = 0; j < 4; ++j) stage_copy16(bgsrc[j] + kk, &Bbuf[st][boff[j]]);
  };

  int nK = K >> 5;
  issue(0, 0);

  v8f acc[2][4] = {};
  for (int i = 0; i < nK; ++i) {
    int st = i & 1;
    bool hasNext = (i + 1) < nK;
    if (hasNext) issue((i + 1) << 5, st ^ 1);
    wait_stage(hasNext);  // current stage resident + visible to all waves

    const bf16* Ab = &Abuf[st][0];
    const bf16* Bb = &Bbuf[st][0];
    v16bf a0 = load_frag_lds(Ab + (wm * 32 + n16) * LDSP, half);
    v16bf a1 = load_frag_lds(Ab + (wm * 32 + 16 + n16) * LDSP, half);
#pragma unroll
    for (int nt = 0; nt < 4; ++nt) {
      v16bf b = load_frag_lds(Bb + (wn * 64 + nt * 16 + n16) * LDSP, half);
      acc[0][nt] = WMMA_BF16(a0, b, acc[0][nt]);
      acc[1][nt] = WMMA_BF16(a1, b, acc[1][nt]);
    }
    __syncthreads();  // everyone done with stage st before it's overwritten
  }

  // C/D layout: VGPR r -> row (r + 8*half), col = lane&15 within a 16x16 tile.
  int row0 = rowblk + wm * 32;
  int col0 = colblk + wn * 64;
#pragma unroll
  for (int mt = 0; mt < 2; ++mt)
#pragma unroll
    for (int nt = 0; nt < 4; ++nt) {
      int col = col0 + nt * 16 + n16;
      if (col >= N) continue;
#pragma unroll
      for (int r = 0; r < 8; ++r) {
        int row = row0 + mt * 16 + r + 8 * half;
        if (row >= M) continue;
        float v = acc[mt][nt][r];
        if (Cf) Cf[(size_t)row * ldc + col] = v;
        if (Cb) Cb[(size_t)row * ldc + col] = (bf16)v;
      }
    }
}

// ---------------------------------------------------------------------------
// RoPE: raw[row][64] -> rotated; optional f32 out (stride) and bf16 out.
// row = b*T + t ; pair d uses inv = 10000^(-d/32).
// ---------------------------------------------------------------------------
__global__ void rope_kernel(const float* __restrict__ raw, float* __restrict__ out_f,
                            int f_stride, bf16* __restrict__ out_b, int Tdim) {
  int row = blockIdx.x;
  int d = threadIdx.x;  // 32 threads
  int t = row % Tdim;
  float inv = __powf(10000.0f, -(float)d / 32.0f);
  float s, c;
  __sincosf((float)t * inv, &s, &c);
  float x1 = raw[(size_t)row * 64 + d];
  float x2 = raw[(size_t)row * 64 + d + 32];
  float o1 = x1 * c - x2 * s;
  float o2 = x2 * c + x1 * s;
  if (out_f) {
    out_f[(size_t)row * f_stride + d] = o1;
    out_f[(size_t)row * f_stride + d + 32] = o2;
  }
  if (out_b) {
    out_b[(size_t)row * 64 + d] = (bf16)o1;
    out_b[(size_t)row * 64 + d + 32] = (bf16)o2;
  }
}

// entry[row][0..511] = ckv_f32
__global__ void pack_entry(const float* __restrict__ ckv, float* __restrict__ entry) {
  long i = (long)blockIdx.x * 256 + threadIdx.x;  // 4096*512
  if (i >= (long)4096 * 512) return;
  int c = (int)(i & 511);
  long row = i >> 9;
  entry[row * 576 + c] = ckv[i];
}

// out[((b*16+h)*2048 + t)*192 + c] : c<128 from full[(b*2048+t)*2048 + h*128+c],
// else from rb[(b*2048+t)*64 + c-128]   (folds rope part into attention GEMM)
__global__ void pack_qk(const bf16* __restrict__ full, const bf16* __restrict__ rb,
                        bf16* __restrict__ out) {
  long i = (long)blockIdx.x * 256 + threadIdx.x;
  long total = (long)32 * 2048 * 192;
  if (i >= total) return;
  int c = (int)(i % 192);
  long rem = i / 192;
  int t = (int)(rem & 2047);
  int bh = (int)(rem >> 11);
  int b = bh >> 4, h = bh & 15;
  bf16 v;
  if (c < 128) v = full[((size_t)(b * 2048 + t)) * 2048 + h * 128 + c];
  else         v = rb[((size_t)(b * 2048 + t)) * 64 + (c - 128)];
  out[i] = v;
}

// vT[((b*16+h)*128 + d)*2048 + t] = vfull[(b*2048+t)*2048 + h*128 + d]
__global__ void pack_vT(const bf16* __restrict__ vfull, bf16* __restrict__ vT) {
  long i = (long)blockIdx.x * 256 + threadIdx.x;
  long total = (long)32 * 128 * 2048;
  if (i >= total) return;
  int t = (int)(i & 2047);
  long rem = i >> 11;
  int d = (int)(rem & 127);
  int bh = (int)(rem >> 7);
  int b = bh >> 4, h = bh & 15;
  vT[i] = vfull[((size_t)(b * 2048 + t)) * 2048 + h * 128 + d];
}

// ---------------------------------------------------------------------------
// Flash attention: scores = q'(192) . k'(192)^T / sqrt(192), causal, online
// softmax, O = P * V.  Block = 8 waves, each wave owns 16 q-rows of one
// (b,h); grid = (B*NH, T/128).  P is relaid out C->A through per-wave LDS.
// ---------------------------------------------------------------------------
__global__ __launch_bounds__(256) void fa_kernel(
    const bf16* __restrict__ qp, const bf16* __restrict__ kp,
    const bf16* __restrict__ vT, bf16* __restrict__ attn) {
  __shared__ __align__(16) bf16 pstage[8][16][32];

  int bh = blockIdx.x;                 // b*16 + h
  int wave = threadIdx.x >> 5, lane = threadIdx.x & 31;
  int half = lane >> 4, n = lane & 15;
  int qt0 = blockIdx.y * 128 + wave * 16;

  const bf16* qrow = qp + ((size_t)bh * 2048 + qt0 + n) * 192;
  v16bf qf[6];
#pragma unroll
  for (int kc = 0; kc < 6; ++kc) qf[kc] = load_frag(qrow, kc * 32, half);

  v8f O[8] = {};
  float Mrun[8], Lrun[8];
#pragma unroll
  for (int r = 0; r < 8; ++r) { Mrun[r] = -INFINITY; Lrun[r] = 0.0f; }

  const float sc = 0.07216878364870322f;  // 1/sqrt(192) = 1/sqrt(HD+DR)

  for (int jb = 0; jb <= qt0 + 15; jb += 32) {
    const bf16* krow0 = kp + ((size_t)bh * 2048 + jb + n) * 192;
    const bf16* krow1 = krow0 + (size_t)16 * 192;
    __builtin_prefetch(krow0 + (size_t)32 * 192, 0, 1);  // next key tile
    v8f s0 = {}, s1 = {};
#pragma unroll
    for (int kc = 0; kc < 6; ++kc) {
      s0 = WMMA_BF16(qf[kc], load_frag(krow0, kc * 32, half), s0);
      s1 = WMMA_BF16(qf[kc], load_frag(krow1, kc * 32, half), s1);
    }

    bool need_mask = (jb + 31) > qt0;
    float f[8];
#pragma unroll
    for (int r = 0; r < 8; ++r) {
      int rowi = qt0 + r + 8 * half;
      float a = s0[r] * sc, b = s1[r] * sc;
      if (need_mask) {
        if (jb + n > rowi) a = -INFINITY;
        if (jb + 16 + n > rowi) b = -INFINITY;
      }
      float vmax = fmaxf(a, b);
#pragma unroll
      for (int m = 1; m < 16; m <<= 1) vmax = fmaxf(vmax, __shfl_xor(vmax, m));
      float mnew = fmaxf(Mrun[r], vmax);
      f[r] = __expf(Mrun[r] - mnew);
      Mrun[r] = mnew;
      float p0 = __expf(a - mnew), p1 = __expf(b - mnew);
      float rs = p0 + p1;
#pragma unroll
      for (int m = 1; m < 16; m <<= 1) rs += __shfl_xor(rs, m);
      Lrun[r] = Lrun[r] * f[r] + rs;
      pstage[wave][r + 8 * half][n] = (bf16)p0;
      pstage[wave][r + 8 * half][16 + n] = (bf16)p1;
    }
#pragma unroll
    for (int dt = 0; dt < 8; ++dt)
#pragma unroll
      for (int r = 0; r < 8; ++r) O[dt][r] *= f[r];

    __builtin_amdgcn_wave_barrier();   // LDS in-order per wave; block reordering
    v16bf pa;
    {
      const bf16* pr = &pstage[wave][n][0];
      v8bf lo = *(const v8bf*)(pr + half * 8);
      v8bf hi = *(const v8bf*)(pr + 16 + half * 8);
#pragma unroll
      for (int i = 0; i < 8; ++i) { pa[i] = lo[i]; pa[i + 8] = hi[i]; }
    }
    __builtin_amdgcn_wave_barrier();

#pragma unroll
    for (int dt = 0; dt < 8; ++dt) {
      const bf16* vrow = vT + ((size_t)bh * 128 + dt * 16 + n) * 2048 + jb;
      v8bf lo = *(const v8bf*)(vrow + half * 8);
      v8bf hi = *(const v8bf*)(vrow + 16 + half * 8);
      v16bf vb;
#pragma unroll
      for (int i = 0; i < 8; ++i) { vb[i] = lo[i]; vb[i + 8] = hi[i]; }
      O[dt] = WMMA_BF16(pa, vb, O[dt]);
    }
  }

  int b = bh >> 4, h = bh & 15;
#pragma unroll
  for (int r = 0; r < 8; ++r) {
    float inv = 1.0f / Lrun[r];
    int t = qt0 + r + 8 * half;
    size_t base = ((size_t)(b * 2048 + t)) * 2048 + h * 128;
#pragma unroll
    for (int dt = 0; dt < 8; ++dt)
      attn[base + dt * 16 + n] = (bf16)(O[dt][r] * inv);
  }
}

// ---------------------------------------------------------------------------
// Host launcher
// ---------------------------------------------------------------------------
extern "C" void kernel_launch(void* const* d_in, const int* in_sizes, int n_in,
                              void* d_out, int out_size, void* d_ws, size_t ws_size,
                              hipStream_t stream) {
  (void)in_sizes; (void)n_in; (void)out_size; (void)ws_size;
  const int B = 2, T = 2048, D = 2048, DC = 512, DR = 64, NH = 16;
  const long M = (long)B * T;              // 4096
  const long OUT0 = (long)B * T * D;       // 8388608 floats (main output)

  const float* x    = (const float*)d_in[0];
  const float* Wkv  = (const float*)d_in[1];
  const float* Wkup = (const float*)d_in[2];
  const float* Wvup = (const float*)d_in[3];
  const float* Wq   = (const float*)d_in[4];
  const float* Wqr  = (const float*)d_in[5];
  const float* Wkr  = (const float*)d_in[6];
  const float* Wo   = (const float*)d_in[7];
  float* out   = (float*)d_out;
  float* entry = (float*)d_out + OUT0;     // [B,T,576]

  // workspace bump allocator (256B aligned)
  size_t off = 0;
  auto alloc = [&](size_t bytes) -> char* {
    off = (off + 255) & ~(size_t)255;
    char* p = (char*)d_ws + off;
    off += bytes;
    return p;
  };
  bf16* xb     = (bf16*)alloc(M * D * 2);
  bf16* Wkv_t  = (bf16*)alloc((size_t)DC * D * 2);
  bf16* Wkup_t = (bf16*)alloc((size_t)D * DC * 2);
  bf16* Wvup_t = (bf16*)alloc((size_t)D * DC * 2);
  bf16* Wq_t   = (bf16*)alloc((size_t)D * D * 2);
  bf16* Wqr_t  = (bf16*)alloc((size_t)DR * D * 2);
  bf16* Wkr_t  = (bf16*)alloc((size_t)DR * DC * 2);
  bf16* Wo_t   = (bf16*)alloc((size_t)D * D * 2);
  float* ckv_f = (float*)alloc(M * DC * 4);
  bf16* ckv_b  = (bf16*)alloc(M * DC * 2);
  float* kr_raw= (float*)alloc(M * DR * 4);
  bf16* kr_b   = (bf16*)alloc(M * DR * 2);
  float* qr_raw= (float*)alloc(M * DR * 4);
  bf16* qr_b   = (bf16*)alloc(M * DR * 2);
  bf16* qfull  = (bf16*)alloc(M * D * 2);
  bf16* kfull  = (bf16*)alloc(M * D * 2);
  bf16* vfull  = (bf16*)alloc(M * D * 2);
  bf16* qprime = (bf16*)alloc((size_t)B * NH * T * 192 * 2);
  bf16* kprime = (bf16*)alloc((size_t)B * NH * T * 192 * 2);
  bf16* vTb    = (bf16*)alloc((size_t)B * NH * 128 * T * 2);
  bf16* attn_b = xb;  // xb is dead by the time fa_kernel runs

  // 1) casts
  cast_f32_bf16<<<dim3((M * D + 255) / 256), 256, 0, stream>>>(x, xb, M * D);
  dim3 tb(32, 8);
  cast_transpose<<<dim3(DC / 32, D / 32), tb, 0, stream>>>(Wkv, Wkv_t, D, DC);
  cast_transpose<<<dim3(D / 32, DC / 32), tb, 0, stream>>>(Wkup, Wkup_t, DC, D);
  cast_transpose<<<dim3(D / 32, DC / 32), tb, 0, stream>>>(Wvup, Wvup_t, DC, D);
  cast_transpose<<<dim3(D / 32, D / 32),  tb, 0, stream>>>(Wq, Wq_t, D, D);
  cast_transpose<<<dim3(DR / 32, D / 32), tb, 0, stream>>>(Wqr, Wqr_t, D, DR);
  cast_transpose<<<dim3(DR / 32, DC / 32),tb, 0, stream>>>(Wkr, Wkr_t, DC, DR);
  cast_transpose<<<dim3(D / 32, D / 32),  tb, 0, stream>>>(Wo, Wo_t, D, D);

  // 2) ckv = x @ Wkv  (f32 for entry, bf16 for downstream)
  gemm_bf16<<<dim3(M / 64, (DC + 255) / 256), 256, 0, stream>>>(
      xb, D, Wkv_t, D, ckv_f, ckv_b, DC, (int)M, DC, D);
  // 3) kr = ckv @ Wkr, then rope -> entry cols 512.. and bf16
  gemm_bf16<<<dim3(M / 64, 1), 256, 0, stream>>>(
      ckv_b, DC, Wkr_t, DC, kr_raw, nullptr, DR, (int)M, DR, DC);
  rope_kernel<<<dim3((int)M), 32, 0, stream>>>(kr_raw, entry + 512, 576, kr_b, T);
  pack_entry<<<dim3((M * DC + 255) / 256), 256, 0, stream>>>(ckv_f, entry);

  // 4) q = x@Wq, qr = rope(x@Wqr)
  gemm_bf16<<<dim3(M / 64, D / 256), 256, 0, stream>>>(
      xb, D, Wq_t, D, nullptr, qfull, D, (int)M, D, D);
  gemm_bf16<<<dim3(M / 64, 1), 256, 0, stream>>>(
      xb, D, Wqr_t, D, qr_raw, nullptr, DR, (int)M, DR, D);
  rope_kernel<<<dim3((int)M), 32, 0, stream>>>(qr_raw, nullptr, 0, qr_b, T);

  // 5) k = ckv@Wkup, v = ckv@Wvup
  gemm_bf16<<<dim3(M / 64, D / 256), 256, 0, stream>>>(
      ckv_b, DC, Wkup_t, DC, nullptr, kfull, D, (int)M, D, DC);
  gemm_bf16<<<dim3(M / 64, D / 256), 256, 0, stream>>>(
      ckv_b, DC, Wvup_t, DC, nullptr, vfull, D, (int)M, D, DC);

  // 6) pack attention operands
  long nqk = (long)B * NH * T * 192;
  pack_qk<<<dim3((nqk + 255) / 256), 256, 0, stream>>>(qfull, qr_b, qprime);
  pack_qk<<<dim3((nqk + 255) / 256), 256, 0, stream>>>(kfull, kr_b, kprime);
  long nvt = (long)B * NH * 128 * T;
  pack_vT<<<dim3((nvt + 255) / 256), 256, 0, stream>>>(vfull, vTb);

  // 7) flash attention
  fa_kernel<<<dim3(B * NH, T / 128), 256, 0, stream>>>(qprime, kprime, vTb, attn_b);

  // 8) out = attn @ Wo  (f32 into d_out)
  gemm_bf16<<<dim3(M / 64, D / 256), 256, 0, stream>>>(
      attn_b, D, Wo_t, D, out, nullptr, D, (int)M, D, D);
}
